// SomaMind_70858370449865
// MI455X (gfx1250) — matrix operand order repaired and checked by reading, not compiled, not run on hardware
//
#include <hip/hip_runtime.h>
#include <hip/hip_bf16.h>
#include <math.h>

// ---------------- problem constants ----------------
constexpr int V = 32000, E = 64, HID = 128, D = 256, OE = 32, NPRIM = 32;
constexpr int START = 32, T = 16, REF = 64, Bb = 512, Ss = 256;
constexpr int ENC2 = 2 * HID;           // 256
constexpr float NEGF = -1000000000.0f;

// ---------------- WMMA helpers ----------------
typedef __attribute__((ext_vector_type(16))) __bf16 bf16x16;
typedef __attribute__((ext_vector_type(8)))  __bf16 bf16x8;
typedef __attribute__((ext_vector_type(8)))  float  f32x8;

union ABFrag { bf16x16 v; bf16x8 h[2]; };

__device__ inline f32x8 wmma_bf16(bf16x16 a, bf16x16 b, f32x8 c) {
  return __builtin_amdgcn_wmma_f32_16x16x32_bf16(false, a, false, b, (short)0, c, false, false);
}

// A fragment: 16x32 (MxK). Row-major A with leading dim ldk (elements).
__device__ inline bf16x16 load_a_frag(const __bf16* A, int ldk, int m0, int k0, int lane) {
  int m = m0 + (lane & 15);
  int hi = lane >> 4;
  const __bf16* base = A + (size_t)m * ldk + k0 + hi * 8;
  ABFrag r;
  r.h[0] = *(const bf16x8*)(base);
  r.h[1] = *(const bf16x8*)(base + 16);
  return r.v;
}

// B fragment: 32x16 (KxN), sourced from Bt = B transposed, row-major (N, ldk).
__device__ inline bf16x16 load_b_frag(const __bf16* Bt, int ldk, int n0, int k0, int lane) {
  int n = n0 + (lane & 15);
  int hi = lane >> 4;
  const __bf16* base = Bt + (size_t)n * ldk + k0 + hi * 8;
  ABFrag r;
  r.h[0] = *(const bf16x8*)(base);
  r.h[1] = *(const bf16x8*)(base + 16);
  return r.v;
}

__device__ inline float sigf(float x) { return 1.0f / (1.0f + __expf(-x)); }

// ---------------- generic bf16 WMMA GEMM ----------------
// C(M,N) = A(M,K) @ Bt(N,K)^T + bias ; optional tanh ; optional bf16 copy out.
// REQUIRES: M % 64 == 0, N % 32 == 0, K % 32 == 0.
__global__ void __launch_bounds__(256)
gemm_bf16_kernel(const __bf16* __restrict__ A, const __bf16* __restrict__ Bt,
                 const float* __restrict__ bias, float* __restrict__ C,
                 __bf16* __restrict__ Cb, int M, int N, int K, int act) {
  int lane = threadIdx.x & 31;
  int wave = __builtin_amdgcn_readfirstlane(threadIdx.x >> 5);  // scalar wave id
  int wm = wave >> 2, wn = wave & 3;
  int m0 = blockIdx.y * 64 + wm * 32;
  int n0 = blockIdx.x * 128 + wn * 32;
  if (m0 >= M || n0 >= N) return;       // scalar (wave-uniform) exit
  f32x8 acc00 = {}, acc01 = {}, acc10 = {}, acc11 = {};
  for (int k0 = 0; k0 < K; k0 += 32) {
    bf16x16 a0 = load_a_frag(A, K, m0, k0, lane);
    bf16x16 a1 = load_a_frag(A, K, m0 + 16, k0, lane);
    bf16x16 b0 = load_b_frag(Bt, K, n0, k0, lane);
    bf16x16 b1 = load_b_frag(Bt, K, n0 + 16, k0, lane);
    acc00 = wmma_bf16(a0, b0, acc00);
    acc10 = wmma_bf16(a1, b0, acc10);
    acc01 = wmma_bf16(a0, b1, acc01);
    acc11 = wmma_bf16(a1, b1, acc11);
  }
  int lc = lane & 15, hi = lane >> 4;
  f32x8 accs[2][2] = {{acc00, acc01}, {acc10, acc11}};
  for (int mi = 0; mi < 2; ++mi)
    for (int ni = 0; ni < 2; ++ni) {
      int nn = n0 + ni * 16 + lc;
      for (int v = 0; v < 8; ++v) {
        int mm = m0 + mi * 16 + v + 8 * hi;
        float val = accs[mi][ni][v];
        if (bias) val += bias[nn];
        if (act == 1) val = tanhf(val);
        size_t off = (size_t)mm * N + nn;
        if (C)  C[off] = val;
        if (Cb) Cb[off] = (__bf16)val;
      }
    }
}

// ---------------- LSTM recurrence (persistent over time, both directions) ----------------
// grid = (B/16, 2). Each block owns 16 batch rows for one direction.
// 512 threads = 16 waves; wave w owns gate columns [w*32, w*32+32).
// Whh^T fragments are preloaded into VGPRs and reused across all 256 timesteps.
// pre_*: (B,S,512) bf16 already holds x@Wih + b.  enc out: (B,S,256), fwd->ch0, bwd->ch128.
__global__ void __launch_bounds__(512)
lstm_recur_kernel(const __bf16* __restrict__ pre_f, const __bf16* __restrict__ pre_b,
                  const __bf16* __restrict__ WhhT_f, const __bf16* __restrict__ WhhT_b,
                  const int* __restrict__ lengths, float* __restrict__ enc) {
  __shared__ __align__(16) float  hC[16][HID];
  __shared__ __align__(16) float  cC[16][HID];
  __shared__ __align__(16) __bf16 hB[16][HID];
  __shared__ __align__(16) float  gates[16][4 * HID];
  int dir = blockIdx.y;                        // 0 = fwd, 1 = bwd
  const __bf16* pre  = dir ? pre_b : pre_f;
  const __bf16* WhhT = dir ? WhhT_b : WhhT_f;
  int ch_off = dir * HID;
  int b0 = blockIdx.x * 16;
  int tid = threadIdx.x;
  int lane = tid & 31;
  int wave = __builtin_amdgcn_readfirstlane(tid >> 5);   // 0..15, scalar
  int n0w = wave * 32;                                   // 32 gate columns per wave

  // preload Whh^T fragments into registers (reused for all timesteps)
  bf16x16 bfrag[2][4];
  for (int nf = 0; nf < 2; ++nf)
    for (int kf = 0; kf < 4; ++kf)
      bfrag[nf][kf] = load_b_frag(WhhT, HID, n0w + nf * 16, kf * 32, lane);

  for (int i = tid; i < 16 * HID; i += 512) {
    (&hC[0][0])[i] = 0.f; (&cC[0][0])[i] = 0.f; (&hB[0][0])[i] = (__bf16)0.f;
  }
  __syncthreads();

  int lc = lane & 15, hi = lane >> 4;
  for (int t = 0; t < Ss; ++t) {
    f32x8 acc0 = {}, acc1 = {};
    #pragma unroll
    for (int kf = 0; kf < 4; ++kf) {
      bf16x16 a = load_a_frag(&hB[0][0], HID, 0, kf * 32, lane);
      acc0 = wmma_bf16(a, bfrag[0][kf], acc0);
      acc1 = wmma_bf16(a, bfrag[1][kf], acc1);
    }
    #pragma unroll
    for (int v = 0; v < 8; ++v) {
      gates[v + 8 * hi][n0w + lc]      = acc0[v];
      gates[v + 8 * hi][n0w + 16 + lc] = acc1[v];
    }
    __syncthreads();
    for (int u = tid; u < 16 * HID; u += 512) {
      int m = u >> 7, j = u & 127;
      int b = b0 + m;
      int len = lengths[b];
      const __bf16* pg = pre + ((size_t)b * Ss + t) * 512;
      float ii = (float)pg[j]       + gates[m][j];
      float ff = (float)pg[j + 128] + gates[m][j + 128];
      float gg = (float)pg[j + 256] + gates[m][j + 256];
      float oo = (float)pg[j + 384] + gates[m][j + 384];
      float c_old = cC[m][j], h_old = hC[m][j];
      float c_n = sigf(ff) * c_old + sigf(ii) * tanhf(gg);
      float h_n = sigf(oo) * tanhf(c_n);
      bool valid = t < len;
      float h_keep = valid ? h_n : h_old;
      float c_keep = valid ? c_n : c_old;
      hC[m][j] = h_keep; cC[m][j] = c_keep; hB[m][j] = (__bf16)h_keep;
      int pos = dir ? (valid ? (len - 1 - t) : t) : t;
      enc[((size_t)b * Ss + pos) * ENC2 + ch_off + j] = valid ? h_n : 0.f;
    }
    __syncthreads();
  }
}

// ---------------- span scores: per-b (T x enc2) @ enc_b^T via WMMA ----------------
__global__ void __launch_bounds__(256)
span_scores_kernel(const __bf16* __restrict__ qsp, const __bf16* __restrict__ encB,
                   const int* __restrict__ lengths, float* __restrict__ out) {
  int blk = blockIdx.x;
  int b = blk >> 1;
  int scol = (blk & 1) * 128;
  int lane = threadIdx.x & 31;
  int wave = __builtin_amdgcn_readfirstlane(threadIdx.x >> 5);
  int s0 = scol + wave * 16;
  const __bf16* A  = qsp  + (size_t)b * T * ENC2;
  const __bf16* Bt = encB + (size_t)b * Ss * ENC2;
  f32x8 acc = {};
  for (int k0 = 0; k0 < ENC2; k0 += 32) {
    bf16x16 a = load_a_frag(A, ENC2, 0, k0, lane);
    bf16x16 bfr = load_b_frag(Bt, ENC2, s0, k0, lane);
    acc = wmma_bf16(a, bfr, acc);
  }
  int lc = lane & 15, hi = lane >> 4;
  int len = lengths[b];
  int s = s0 + lc;
  for (int v = 0; v < 8; ++v) {
    int tt = v + 8 * hi;
    out[((size_t)b * T + tt) * Ss + s] = (s < len) ? acc[v] : NEGF;
  }
}

// ---------------- small helper kernels ----------------
__global__ void transpose_bf16_kernel(const float* __restrict__ src, __bf16* __restrict__ dst,
                                      int K, int N) {
  int idx = blockIdx.x * 256 + threadIdx.x;
  if (idx >= K * N) return;
  int k = idx / N, n = idx % N;
  dst[(size_t)n * K + k] = (__bf16)src[idx];
}

__global__ void embed_gather_kernel(const float* __restrict__ emb, const int* __restrict__ ids,
                                    const int* __restrict__ lengths, __bf16* __restrict__ dst,
                                    int rev) {
  int idx = blockIdx.x * 256 + threadIdx.x;   // B*S*E
  if (idx >= Bb * Ss * E) return;
  int e = idx & 63, t = (idx >> 6) & 255, b = idx >> 14;
  int len = lengths[b];
  int ts = rev ? ((t < len) ? (len - 1 - t) : t) : t;
  int id = ids[b * Ss + ts];
  dst[idx] = (__bf16)emb[(size_t)id * E + e];
}

__global__ void gather_conv_kernel(const float* __restrict__ src, const int* __restrict__ lengths,
                                   __bf16* __restrict__ dst, int rev) {
  size_t idx = (size_t)blockIdx.x * 256 + threadIdx.x;  // B*S*256
  int d = (int)(idx & 255), t = (int)((idx >> 8) & 255), b = (int)(idx >> 16);
  int len = lengths[b];
  int ts = rev ? ((t < len) ? (len - 1 - t) : t) : t;
  dst[idx] = (__bf16)src[((size_t)b * Ss + ts) * ENC2 + d];
}

__global__ void pool_kernel(const float* __restrict__ enc, const int* __restrict__ lengths,
                            float* __restrict__ pooled, __bf16* __restrict__ pooledB) {
  int b = blockIdx.x, d = threadIdx.x;
  int len = lengths[b];
  float s = 0.f;
  const float* e = enc + (size_t)b * Ss * ENC2 + d;
  for (int t = 0; t < len; ++t) s += e[(size_t)t * ENC2];
  float v = s / (float)len;
  pooled[b * ENC2 + d] = v;
  pooledB[b * ENC2 + d] = (__bf16)v;
}

__global__ void attn_softmax_kernel(const float* __restrict__ q, const float* __restrict__ enc,
                                    const int* __restrict__ lengths, float* __restrict__ w) {
  __shared__ float qs[ENC2];
  __shared__ float red[Ss];
  int b = blockIdx.x, tid = threadIdx.x;
  qs[tid] = q[b * ENC2 + tid];
  __syncthreads();
  const float* e = enc + ((size_t)b * Ss + tid) * ENC2;
  float sc = 0.f;
  for (int d = 0; d < ENC2; ++d) sc += qs[d] * e[d];
  int len = lengths[b];
  float m = (tid < len) ? sc : NEGF;
  red[tid] = m; __syncthreads();
  for (int st = 128; st > 0; st >>= 1) {
    if (tid < st) red[tid] = fmaxf(red[tid], red[tid + st]);
    __syncthreads();
  }
  float mx = red[0]; __syncthreads();
  float ex = __expf(m - mx);
  red[tid] = ex; __syncthreads();
  for (int st = 128; st > 0; st >>= 1) {
    if (tid < st) red[tid] += red[tid + st];
    __syncthreads();
  }
  w[b * Ss + tid] = ex / red[0];
}

__global__ void attn_ctx_kernel(const float* __restrict__ w, const float* __restrict__ enc,
                                float* __restrict__ ctx) {
  int b = blockIdx.x, d = threadIdx.x;
  float s = 0.f;
  const float* e = enc + (size_t)b * Ss * ENC2 + d;
  const float* wb = w + b * Ss;
  for (int t = 0; t < Ss; ++t) s += wb[t] * e[(size_t)t * ENC2];
  ctx[b * ENC2 + d] = s;
}

__global__ void build_gin_kernel(const int* __restrict__ tgt, const float* __restrict__ op_emb,
                                 const float* __restrict__ ctx, __bf16* __restrict__ gin, int t) {
  int b = blockIdx.x, i = threadIdx.x;   // 288 threads
  int op = (t == 0) ? START : tgt[b * T + (t - 1)];
  float v = (i < OE) ? op_emb[op * OE + i] : ctx[b * ENC2 + (i - OE)];
  gin[(size_t)b * 288 + i] = (__bf16)v;
}

__global__ void gru_cell_kernel(const float* __restrict__ gi, const float* __restrict__ gh,
                                float* __restrict__ h, __bf16* __restrict__ hB,
                                __bf16* __restrict__ hsB, int t) {
  int b = blockIdx.x, d = threadIdx.x;   // 256
  float ri = gi[b * 768 + d], zi = gi[b * 768 + 256 + d], ni = gi[b * 768 + 512 + d];
  float rh = gh[b * 768 + d], zh = gh[b * 768 + 256 + d], nh = gh[b * 768 + 512 + d];
  float r = sigf(ri + rh), z = sigf(zi + zh);
  float n = tanhf(ni + r * nh);
  float hn = (1.f - z) * n + z * h[b * 256 + d];
  h[b * 256 + d] = hn;
  hB[b * 256 + d] = (__bf16)hn;
  hsB[((size_t)b * T + t) * 256 + d] = (__bf16)hn;
}

__global__ void ref_scores_kernel(const float* __restrict__ q, const float* __restrict__ k,
                                  float* __restrict__ out) {
  int idx = blockIdx.x * 256 + threadIdx.x;   // B*T*T
  if (idx >= Bb * T * T) return;
  int s = idx & 15, tt = (idx >> 4) & 15, b = idx >> 8;
  float sc = 0.f;
  const float* qp = q + ((size_t)b * T + tt) * REF;
  const float* kp = k + ((size_t)b * T + s) * REF;
  for (int r = 0; r < REF; ++r) sc += qp[r] * kp[r];
  out[idx] = (tt > s) ? sc : NEGF;
}

// ---------------- host side ----------------
static inline char* wsal(char*& cur, size_t bytes) {
  char* p = cur;
  cur += (bytes + 255) & ~(size_t)255;
  return p;
}

static void launch_gemm(const __bf16* A, const __bf16* Bt, const float* bias,
                        float* C, __bf16* Cb, int M, int N, int K, int act,
                        hipStream_t s) {
  dim3 grid((N + 127) / 128, M / 64);
  gemm_bf16_kernel<<<grid, 256, 0, s>>>(A, Bt, bias, C, Cb, M, N, K, act);
}

static void xpose(const void* src, __bf16* dst, int K, int N, hipStream_t s) {
  int n = K * N;
  transpose_bf16_kernel<<<(n + 255) / 256, 256, 0, s>>>((const float*)src, dst, K, N);
}

extern "C" void kernel_launch(void* const* d_in, const int* in_sizes, int n_in,
                              void* d_out, int out_size, void* d_ws, size_t ws_size,
                              hipStream_t stream) {
  // ---- inputs (setup_inputs dict order, depth-first) ----
  const int*   ids  = (const int*)d_in[0];
  const int*   len  = (const int*)d_in[1];
  const int*   tgt  = (const int*)d_in[2];
  const float* emb  = (const float*)d_in[3];
  // lstm: [0].f (4,5,6) [0].b (7,8,9) [1].f (10,11,12) [1].b (13,14,15)
  const float* op_emb = (const float*)d_in[16];
  float* out = (float*)d_out;

  // ---- workspace carve-up ----
  char* cur = (char*)d_ws;
  __bf16* XG   = (__bf16*)wsal(cur, (size_t)Bb * Ss * 256 * 2);
  __bf16* PREF = (__bf16*)wsal(cur, (size_t)Bb * Ss * 512 * 2);  // fwd pregates (bf16)
  __bf16* PREB = (__bf16*)wsal(cur, (size_t)Bb * Ss * 512 * 2);  // bwd pregates (bf16)
  float*  ENC1 = (float*)wsal(cur, (size_t)Bb * Ss * 256 * 4);
  float*  ENCO = (float*)wsal(cur, (size_t)Bb * Ss * 256 * 4);
  __bf16* ENCB = (__bf16*)wsal(cur, (size_t)Bb * Ss * 256 * 2);
  __bf16* wihT[4]; __bf16* whhT[4];
  int wihK[4] = {E, E, ENC2, ENC2};
  for (int i = 0; i < 4; ++i) {
    wihT[i] = (__bf16*)wsal(cur, (size_t)512 * wihK[i] * 2);
    whhT[i] = (__bf16*)wsal(cur, (size_t)512 * HID * 2);
  }
  __bf16* initT = (__bf16*)wsal(cur, 256 * 256 * 2);
  __bf16* cqT   = (__bf16*)wsal(cur, 256 * 256 * 2);
  __bf16* gWihT = (__bf16*)wsal(cur, 768 * 288 * 2);
  __bf16* gWhhT = (__bf16*)wsal(cur, 768 * 256 * 2);
  __bf16* opT   = (__bf16*)wsal(cur, 32 * 256 * 2);
  __bf16* spanT[4];
  for (int i = 0; i < 4; ++i) spanT[i] = (__bf16*)wsal(cur, 256 * 256 * 2);
  __bf16* refT[4];
  for (int i = 0; i < 4; ++i) refT[i] = (__bf16*)wsal(cur, 64 * 256 * 2);
  float*  POOL  = (float*)wsal(cur, Bb * 256 * 4);
  __bf16* POOLB = (__bf16*)wsal(cur, Bb * 256 * 2);
  float*  H     = (float*)wsal(cur, Bb * 256 * 4);
  __bf16* HB    = (__bf16*)wsal(cur, Bb * 256 * 2);
  float*  Q     = (float*)wsal(cur, Bb * 256 * 4);
  float*  WATT  = (float*)wsal(cur, Bb * Ss * 4);
  float*  CTX   = (float*)wsal(cur, Bb * 256 * 4);
  __bf16* GINB  = (__bf16*)wsal(cur, Bb * 288 * 2);
  float*  GI    = (float*)wsal(cur, Bb * 768 * 4);
  float*  GH    = (float*)wsal(cur, Bb * 768 * 4);
  __bf16* HSB   = (__bf16*)wsal(cur, (size_t)Bb * T * 256 * 2);
  __bf16* QSPB  = (__bf16*)wsal(cur, (size_t)Bb * T * 256 * 2);
  float*  QR    = (float*)wsal(cur, (size_t)Bb * T * 64 * 4);
  float*  KR    = (float*)wsal(cur, (size_t)Bb * T * 64 * 4);

  // ---- weight transposes (f32 (K,N) -> bf16 (N,K)) ----
  int wihIdx[4] = {4, 7, 10, 13};
  int whhIdx[4] = {5, 8, 11, 14};
  for (int i = 0; i < 4; ++i) {
    xpose(d_in[wihIdx[i]], wihT[i], wihK[i], 512, stream);
    xpose(d_in[whhIdx[i]], whhT[i], HID, 512, stream);
  }
  xpose(d_in[17], initT, 256, 256, stream);
  xpose(d_in[19], cqT,   256, 256, stream);
  xpose(d_in[21], gWihT, 288, 768, stream);
  xpose(d_in[22], gWhhT, 256, 768, stream);
  xpose(d_in[25], opT,   256, 32,  stream);
  int spanW[4] = {27, 29, 31, 33};
  for (int i = 0; i < 4; ++i) xpose(d_in[spanW[i]], spanT[i], 256, 256, stream);
  int refW[4] = {35, 37, 39, 41};
  for (int i = 0; i < 4; ++i) xpose(d_in[refW[i]], refT[i], 256, 64, stream);

  const int MBS = Bb * Ss;
  int embGrid = (Bb * Ss * E + 255) / 256;
  int gcGrid  = (Bb * Ss * 256) / 256;
  dim3 recGrid(Bb / 16, 2);

  // ---- encoder layer 1 (fwd+bwd pregates, then joint recurrence) ----
  embed_gather_kernel<<<embGrid, 256, 0, stream>>>(emb, ids, len, XG, 0);
  launch_gemm(XG, wihT[0], (const float*)d_in[6], nullptr, PREF, MBS, 512, E, 0, stream);
  embed_gather_kernel<<<embGrid, 256, 0, stream>>>(emb, ids, len, XG, 1);
  launch_gemm(XG, wihT[1], (const float*)d_in[9], nullptr, PREB, MBS, 512, E, 0, stream);
  lstm_recur_kernel<<<recGrid, 512, 0, stream>>>(PREF, PREB, whhT[0], whhT[1], len, ENC1);

  // ---- encoder layer 2 ----
  gather_conv_kernel<<<gcGrid, 256, 0, stream>>>(ENC1, len, XG, 0);
  launch_gemm(XG, wihT[2], (const float*)d_in[12], nullptr, PREF, MBS, 512, ENC2, 0, stream);
  gather_conv_kernel<<<gcGrid, 256, 0, stream>>>(ENC1, len, XG, 1);
  launch_gemm(XG, wihT[3], (const float*)d_in[15], nullptr, PREB, MBS, 512, ENC2, 0, stream);
  lstm_recur_kernel<<<recGrid, 512, 0, stream>>>(PREF, PREB, whhT[2], whhT[3], len, ENCO);
  gather_conv_kernel<<<gcGrid, 256, 0, stream>>>(ENCO, len, ENCB, 0);

  // ---- pooled init hidden ----
  pool_kernel<<<Bb, 256, 0, stream>>>(ENCO, len, POOL, POOLB);
  launch_gemm(POOLB, initT, (const float*)d_in[18], H, HB, Bb, 256, 256, 1, stream);

  // ---- decoder: T sequential steps ----
  for (int t = 0; t < T; ++t) {
    launch_gemm(HB, cqT, (const float*)d_in[20], Q, nullptr, Bb, 256, 256, 0, stream);
    attn_softmax_kernel<<<Bb, Ss, 0, stream>>>(Q, ENCO, len, WATT);
    attn_ctx_kernel<<<Bb, 256, 0, stream>>>(WATT, ENCO, CTX);
    build_gin_kernel<<<Bb, 288, 0, stream>>>(tgt, op_emb, CTX, GINB, t);
    launch_gemm(GINB, gWihT, (const float*)d_in[23], GI, nullptr, Bb, 768, 288, 0, stream);
    launch_gemm(HB, gWhhT, (const float*)d_in[24], GH, nullptr, Bb, 768, 256, 0, stream);
    gru_cell_kernel<<<Bb, 256, 0, stream>>>(GI, GH, H, HB, HSB, t);
  }

  // ---- output heads ----
  size_t off_span[4];
  off_span[0] = (size_t)Bb * T * NPRIM;
  for (int i = 1; i < 4; ++i) off_span[i] = off_span[i - 1] + (size_t)Bb * T * Ss;
  size_t off_ref0 = off_span[3] + (size_t)Bb * T * Ss;
  size_t off_ref1 = off_ref0 + (size_t)Bb * T * T;

  launch_gemm(HSB, opT, (const float*)d_in[26], out, nullptr, Bb * T, NPRIM, 256, 0, stream);

  int spanB[4] = {28, 30, 32, 34};
  for (int i = 0; i < 4; ++i) {
    launch_gemm(HSB, spanT[i], (const float*)d_in[spanB[i]], nullptr, QSPB, Bb * T, 256, 256, 0, stream);
    span_scores_kernel<<<Bb * 2, 256, 0, stream>>>(QSPB, ENCB, len, out + off_span[i]);
  }

  launch_gemm(HSB, refT[0], (const float*)d_in[36], QR, nullptr, Bb * T, REF, 256, 0, stream);
  launch_gemm(HSB, refT[1], (const float*)d_in[38], KR, nullptr, Bb * T, REF, 256, 0, stream);
  ref_scores_kernel<<<(Bb * T * T) / 256, 256, 0, stream>>>(QR, KR, out + off_ref0);
  launch_gemm(HSB, refT[2], (const float*)d_in[40], QR, nullptr, Bb * T, REF, 256, 0, stream);
  launch_gemm(HSB, refT[3], (const float*)d_in[42], KR, nullptr, Bb * T, REF, 256, 0, stream);
  ref_scores_kernel<<<(Bb * T * T) / 256, 256, 0, stream>>>(QR, KR, out + off_ref1);
}
